// LlamaAttentionWrapperQK_84997402788416
// MI455X (gfx1250) — matrix operand order repaired and checked by reading, not compiled, not run on hardware
//
#include <hip/hip_runtime.h>
#include <math.h>

typedef __bf16 bf16_t;
typedef __attribute__((ext_vector_type(16))) __bf16 bf16x16;
typedef __attribute__((ext_vector_type(8)))  float  floatx8;

#define HID 4096
#define SEQ 2048
#define NB  2
#define NH  32
#define HD  128
#define NEG_INF -1000000000.0f

union Frag { bf16x16 v; uint4 u[2]; };

__device__ __forceinline__ floatx8 wmma_bf16(bf16x16 a, bf16x16 b, floatx8 c) {
  // D = A(16x32 bf16) * B(32x16 bf16) + C(16x16 f32)
  return __builtin_amdgcn_wmma_f32_16x16x32_bf16(false, a, false, b, (short)0, c, false, false);
}

// ---------------------------------------------------------------- convert
__global__ __launch_bounds__(256) void f32_to_bf16_kernel(
    const float* __restrict__ in, bf16_t* __restrict__ out, int n) {
  int i = (blockIdx.x * 256 + threadIdx.x) * 4;
  if (i >= n) return;
  float4 f = *(const float4*)(in + i);
  union { bf16_t h[4]; uint2 u; } o;
  o.h[0] = (bf16_t)f.x; o.h[1] = (bf16_t)f.y;
  o.h[2] = (bf16_t)f.z; o.h[3] = (bf16_t)f.w;
  *(uint2*)(out + i) = o.u;
}

// ---------------------------------------------------------------- GEMM
// C[row, col] = sum_k X[row,k] * W[col,k]   (X bf16, W fp32 converted inline)
// mode 0: out bf16 at Q/K layout [b,h,s,d]
// mode 1: out bf16 at Vt layout  [b,h,d,s]
// mode 2: out f32 row-major [row, col]
__global__ __launch_bounds__(256) void gemm_kernel(
    const bf16_t* __restrict__ X, const float* __restrict__ W,
    bf16_t* __restrict__ outb, float* __restrict__ outf, int mode) {
  __shared__ __align__(16) bf16_t sA[128][40];
  __shared__ __align__(16) bf16_t sB[128][40];

  const int t    = threadIdx.x;
  const int w    = t >> 5, lane = t & 31;
  const int half = lane >> 4, l16 = lane & 15;
  const int wm   = (w >> 1) * 32, wn = (w & 1) * 64;
  const int m0   = blockIdx.y * 128, n0 = blockIdx.x * 128;
  const int lr   = t >> 1, lc = (t & 1) * 16;

  floatx8 acc[2][4];
#pragma unroll
  for (int mt = 0; mt < 2; ++mt)
#pragma unroll
    for (int nt = 0; nt < 4; ++nt)
#pragma unroll
      for (int r = 0; r < 8; ++r) acc[mt][nt][r] = 0.f;

  for (int k0 = 0; k0 < HID; k0 += 32) {
    __syncthreads();
    { // X tile: 128 rows x 32 k (bf16)
      const uint4* src = (const uint4*)(X + (size_t)(m0 + lr) * HID + k0 + lc);
      uint4 a0 = src[0], a1 = src[1];
      *(uint4*)&sA[lr][lc]     = a0;
      *(uint4*)&sA[lr][lc + 8] = a1;
    }
    { // W tile: 128 cols x 32 k, fp32 -> bf16 inline
      const float4* src = (const float4*)(W + (size_t)(n0 + lr) * HID + k0 + lc);
      float4 f0 = src[0], f1 = src[1], f2 = src[2], f3 = src[3];
      union { bf16_t h[8]; uint4 u; } p0, p1;
      p0.h[0]=(bf16_t)f0.x; p0.h[1]=(bf16_t)f0.y; p0.h[2]=(bf16_t)f0.z; p0.h[3]=(bf16_t)f0.w;
      p0.h[4]=(bf16_t)f1.x; p0.h[5]=(bf16_t)f1.y; p0.h[6]=(bf16_t)f1.z; p0.h[7]=(bf16_t)f1.w;
      p1.h[0]=(bf16_t)f2.x; p1.h[1]=(bf16_t)f2.y; p1.h[2]=(bf16_t)f2.z; p1.h[3]=(bf16_t)f2.w;
      p1.h[4]=(bf16_t)f3.x; p1.h[5]=(bf16_t)f3.y; p1.h[6]=(bf16_t)f3.z; p1.h[7]=(bf16_t)f3.w;
      *(uint4*)&sB[lr][lc]     = p0.u;
      *(uint4*)&sB[lr][lc + 8] = p1.u;
    }
    if (k0 + 32 < HID) { // hint next stage into cache (global_prefetch_b8)
      __builtin_prefetch(X + (size_t)(m0 + lr) * HID + k0 + 32 + lc, 0, 0);
      __builtin_prefetch(W + (size_t)(n0 + lr) * HID + k0 + 32 + lc, 0, 0);
    }
    __syncthreads();

    Frag a[2], b[4];
#pragma unroll
    for (int mt = 0; mt < 2; ++mt) {
      int row = wm + mt * 16 + l16;
      a[mt].u[0] = *(const uint4*)&sA[row][8 * half];
      a[mt].u[1] = *(const uint4*)&sA[row][16 + 8 * half];
    }
#pragma unroll
    for (int nt = 0; nt < 4; ++nt) {
      int col = wn + nt * 16 + l16;
      b[nt].u[0] = *(const uint4*)&sB[col][8 * half];
      b[nt].u[1] = *(const uint4*)&sB[col][16 + 8 * half];
    }
#pragma unroll
    for (int mt = 0; mt < 2; ++mt)
#pragma unroll
      for (int nt = 0; nt < 4; ++nt)
        acc[mt][nt] = wmma_bf16(a[mt].v, b[nt].v, acc[mt][nt]);
  }

  // epilogue
#pragma unroll
  for (int mt = 0; mt < 2; ++mt)
#pragma unroll
    for (int nt = 0; nt < 4; ++nt)
#pragma unroll
      for (int r = 0; r < 8; ++r) {
        int row = m0 + wm + mt * 16 + r + 8 * half;
        int col = n0 + wn + nt * 16 + l16;
        float v = acc[mt][nt][r];
        if (mode == 2) {
          outf[(size_t)row * HID + col] = v;
        } else {
          int b_ = row >> 11, s = row & (SEQ - 1);
          int h  = col >> 7,  d = col & (HD - 1);
          size_t idx;
          if (mode == 0) idx = (((size_t)(b_ * NH + h)) * SEQ + s) * HD + d;
          else           idx = (((size_t)(b_ * NH + h)) * HD + d) * SEQ + s;
          outb[idx] = (bf16_t)v;
        }
      }
}

// ---------------------------------------------------------------- RoPE (in place on Q and K, layout [b,h,s,d])
__global__ __launch_bounds__(256) void rope_kernel(
    bf16_t* __restrict__ Q, bf16_t* __restrict__ K, const int* __restrict__ pos) {
  size_t idx = (size_t)blockIdx.x * 256 + threadIdx.x;
  int d     = (int)(idx & 63);
  size_t r1 = idx >> 6;
  int s     = (int)(r1 & (SEQ - 1));
  size_t bh = r1 >> 11;
  float p   = (float)pos[s];
  // inv_freq = 10000^(-(2d)/128)
  float inv = exp2f(-((float)(2 * d) / 128.0f) * 13.287712379549449f);
  float f   = p * inv;
  float cs  = cosf(f), sn = sinf(f);
  size_t base = (bh * SEQ + s) * HD + d;
  float q1 = (float)Q[base], q2 = (float)Q[base + 64];
  Q[base]      = (bf16_t)(q1 * cs - q2 * sn);
  Q[base + 64] = (bf16_t)(q2 * cs + q1 * sn);
  float k1 = (float)K[base], k2 = (float)K[base + 64];
  K[base]      = (bf16_t)(k1 * cs - k2 * sn);
  K[base + 64] = (bf16_t)(k2 * cs + k1 * sn);
}

// ---------------------------------------------------------------- MXFP4 fake-quant on Vt [b,h,d,s]; blocks = 128 contiguous s
__global__ __launch_bounds__(256) void quant_kernel(bf16_t* __restrict__ Vt, size_t total) {
  size_t gid = (size_t)blockIdx.x * 256 + threadIdx.x;
  int lane = (int)(gid & 31);
  size_t wid = gid >> 5;
  size_t base = wid * 128 + (size_t)lane * 4;
  if (base + 3 >= total) return;
  float x[4];
#pragma unroll
  for (int i = 0; i < 4; ++i) x[i] = (float)Vt[base + i];
  float amax = 0.f;
#pragma unroll
  for (int i = 0; i < 4; ++i) amax = fmaxf(amax, fabsf(x[i]));
#pragma unroll
  for (int m = 1; m < 32; m <<= 1) amax = fmaxf(amax, __shfl_xor(amax, m, 32));
  float scale = exp2f(ceilf(log2f(fmaxf(amax, 1e-30f))) - 3.0f);
  if (amax == 0.f) scale = 1.f;
#pragma unroll
  for (int i = 0; i < 4; ++i) {
    float t = x[i] / scale;
    float a = fabsf(t);
    float e = fminf(fmaxf(floorf(log2f(fmaxf(a, 1e-30f))), 0.f), 2.f);
    float step = exp2f(e) * 0.5f;
    float q = fminf(rintf(a / step) * step, 6.f);
    q = copysignf(q, t);
    Vt[base + i] = (bf16_t)(q * scale);
  }
}

// ---------------------------------------------------------------- Flash attention
// Q,K: [b,h,s,d] bf16 (roped); Vt: [b,h,d,s] bf16 (quantized); O: [b,s,h*d] bf16
__global__ __launch_bounds__(256) void flash_kernel(
    const bf16_t* __restrict__ Q, const bf16_t* __restrict__ K,
    const bf16_t* __restrict__ Vt, bf16_t* __restrict__ O) {
  __shared__ __align__(16) bf16_t kbuf[32][136];      // [key][d]
  __shared__ __align__(16) bf16_t vbuf[128][40];      // [d][s-offset]
  __shared__ __align__(16) bf16_t pbuf[8][16][40];    // per-wave P tile

  const int t    = threadIdx.x;
  const int w    = t >> 5, lane = t & 31;
  const int half = lane >> 4, l16 = lane & 15;
  const int qb   = blockIdx.x, h = blockIdx.y, b = blockIdx.z;
  const size_t bh = (size_t)b * NH + h;
  const int q0   = qb * 128 + w * 16;

  // Q fragments (A layout), 16 rows x 128 d = 4 K-steps
  Frag qf[4];
  {
    const bf16_t* qrow = Q + (bh * SEQ + q0 + l16) * HD;
#pragma unroll
    for (int kk = 0; kk < 4; ++kk) {
      qf[kk].u[0] = *(const uint4*)(qrow + kk * 32 + 8 * half);
      qf[kk].u[1] = *(const uint4*)(qrow + kk * 32 + 16 + 8 * half);
    }
  }

  float m[8], l[8];
  floatx8 o[8];
#pragma unroll
  for (int r = 0; r < 8; ++r) { m[r] = -__builtin_inff(); l[r] = 0.f; }
#pragma unroll
  for (int dt = 0; dt < 8; ++dt)
#pragma unroll
    for (int r = 0; r < 8; ++r) o[dt][r] = 0.f;

  const float scale = 0.08838834764831845f;  // 1/sqrt(128)
  const int jend = (qb + 1) * 128;
  for (int j = 0; j < jend; j += 32) {
    __syncthreads();
    {
      int kr = t >> 3, kc = (t & 7) * 16;
      const uint4* src = (const uint4*)(K + (bh * SEQ + j + kr) * HD + kc);
      *(uint4*)&kbuf[kr][kc]     = src[0];
      *(uint4*)&kbuf[kr][kc + 8] = src[1];
      int vr = t >> 1, vc = (t & 1) * 16;
      const uint4* vsrc = (const uint4*)(Vt + (bh * HD + vr) * SEQ + j + vc);
      *(uint4*)&vbuf[vr][vc]     = vsrc[0];
      *(uint4*)&vbuf[vr][vc + 8] = vsrc[1];
    }
    __syncthreads();

    // S = Q * K^T for 32 keys (2 N-tiles, 4 K-steps over d)
    floatx8 sacc[2];
#pragma unroll
    for (int nt = 0; nt < 2; ++nt)
#pragma unroll
      for (int r = 0; r < 8; ++r) sacc[nt][r] = 0.f;
#pragma unroll
    for (int nt = 0; nt < 2; ++nt)
#pragma unroll
      for (int kk = 0; kk < 4; ++kk) {
        Frag bfr;
        bfr.u[0] = *(const uint4*)&kbuf[nt * 16 + l16][kk * 32 + 8 * half];
        bfr.u[1] = *(const uint4*)&kbuf[nt * 16 + l16][kk * 32 + 16 + 8 * half];
        sacc[nt] = wmma_bf16(qf[kk].v, bfr.v, sacc[nt]);
      }

    // online softmax update
    float p[2][8];
#pragma unroll
    for (int r = 0; r < 8; ++r) {
      int qg = q0 + r + 8 * half;
      float s0 = sacc[0][r] * scale;
      float s1 = sacc[1][r] * scale;
      if (j + l16 > qg)      s0 = NEG_INF;
      if (j + 16 + l16 > qg) s1 = NEG_INF;
      float mc = fmaxf(s0, s1);
#pragma unroll
      for (int d_ = 1; d_ < 16; d_ <<= 1) mc = fmaxf(mc, __shfl_xor(mc, d_, 32));
      float mn = fmaxf(m[r], mc);
      float alpha = __expf(m[r] - mn);
      float p0 = __expf(s0 - mn);
      float p1 = __expf(s1 - mn);
      float rs = p0 + p1;
#pragma unroll
      for (int d_ = 1; d_ < 16; d_ <<= 1) rs += __shfl_xor(rs, d_, 32);
      l[r] = l[r] * alpha + rs;
      m[r] = mn;
      p[0][r] = p0; p[1][r] = p1;
#pragma unroll
      for (int dt = 0; dt < 8; ++dt) o[dt][r] *= alpha;
    }

    // P (C layout) -> per-wave LDS -> A layout fragment
#pragma unroll
    for (int nt = 0; nt < 2; ++nt)
#pragma unroll
      for (int r = 0; r < 8; ++r)
        pbuf[w][r + 8 * half][nt * 16 + l16] = (bf16_t)p[nt][r];

    Frag pf;
    pf.u[0] = *(const uint4*)&pbuf[w][l16][8 * half];
    pf.u[1] = *(const uint4*)&pbuf[w][l16][16 + 8 * half];

    // O += P * V  (8 d-tiles)
#pragma unroll
    for (int dt = 0; dt < 8; ++dt) {
      Frag vf;
      vf.u[0] = *(const uint4*)&vbuf[dt * 16 + l16][8 * half];
      vf.u[1] = *(const uint4*)&vbuf[dt * 16 + l16][16 + 8 * half];
      o[dt] = wmma_bf16(pf.v, vf.v, o[dt]);
    }
  }

  // normalize + store O[b, s, h*128 + d]
#pragma unroll
  for (int r = 0; r < 8; ++r) {
    float inv = 1.f / l[r];
    int qg = q0 + r + 8 * half;
    size_t base = ((size_t)b * SEQ + qg) * HID + (size_t)h * HD;
#pragma unroll
    for (int dt = 0; dt < 8; ++dt)
      O[base + dt * 16 + l16] = (bf16_t)(o[dt][r] * inv);
  }
}

// ---------------------------------------------------------------- launch
extern "C" void kernel_launch(void* const* d_in, const int* in_sizes, int n_in,
                              void* d_out, int out_size, void* d_ws, size_t ws_size,
                              hipStream_t stream) {
  const float* hidden = (const float*)d_in[0];
  const int*   pos    = (const int*)d_in[1];
  const float* Wq     = (const float*)d_in[2];
  const float* Wk     = (const float*)d_in[3];
  const float* Wv     = (const float*)d_in[4];
  const float* Wo     = (const float*)d_in[5];
  float* out = (float*)d_out;

  const size_t EL = (size_t)NB * SEQ * HID;  // 16,777,216 elements
  char* ws = (char*)d_ws;
  bf16_t* Xb = (bf16_t*)(ws);
  bf16_t* Qb = (bf16_t*)(ws + EL * 2);
  bf16_t* Kb = (bf16_t*)(ws + EL * 4);
  bf16_t* Vt = (bf16_t*)(ws + EL * 6);
  bf16_t* Ob = (bf16_t*)(ws + EL * 8);

  f32_to_bf16_kernel<<<(int)(EL / 1024), 256, 0, stream>>>(hidden, Xb, (int)EL);

  dim3 gg(HID / 128, (NB * SEQ) / 128);  // 32 x 32
  gemm_kernel<<<gg, 256, 0, stream>>>(Xb, Wq, Qb, nullptr, 0);
  gemm_kernel<<<gg, 256, 0, stream>>>(Xb, Wk, Kb, nullptr, 0);
  gemm_kernel<<<gg, 256, 0, stream>>>(Xb, Wv, Vt, nullptr, 1);

  rope_kernel<<<(int)((size_t)NB * NH * SEQ * 64 / 256), 256, 0, stream>>>(Qb, Kb, pos);
  quant_kernel<<<(int)(EL / 128 / 8), 256, 0, stream>>>(Vt, EL);

  flash_kernel<<<dim3(SEQ / 128, NH, NB), 256, 0, stream>>>(Qb, Kb, Vt, Ob);

  gemm_kernel<<<gg, 256, 0, stream>>>(Ob, Wo, nullptr, out, 2);
}
